// DeepSeekTrainable_19224273617630
// MI455X (gfx1250) — matrix operand order repaired and checked by reading, not compiled, not run on hardware
//
#include <hip/hip_runtime.h>
#include <math.h>

// ---------------- model constants ----------------
#define S_LEN 2048
#define NLAYER 2
#define DIM 2048
#define NHEAD 16
#define DN 128
#define DR 64
#define DV 128
#define QKD 192           // DN + DR
#define KVR 512
#define INTER 8192
#define VOCAB 32000
#define ATTN_SCALE 0.07216878364870323f   // 192^-0.5

typedef __bf16 bf16;
typedef __attribute__((ext_vector_type(16))) __bf16 v16bf;
typedef __attribute__((ext_vector_type(8)))  float  v8f;
typedef __attribute__((ext_vector_type(4)))  float  v4f;
typedef __attribute__((ext_vector_type(4)))  int    v4i;

static_assert(sizeof(bf16) == 2, "bf16 must be 2 bytes");

// ---------------- gfx1250 async global->LDS copy (ASYNCcnt path) ----------------
#if defined(__has_builtin)
#  if __has_builtin(__builtin_amdgcn_global_load_async_to_lds_b128) && \
      __has_builtin(__builtin_amdgcn_s_wait_asynccnt)
#    define USE_ASYNC_COPY 1
#  endif
#endif
#ifndef USE_ASYNC_COPY
#  define USE_ASYNC_COPY 0
#endif

#if USE_ASYNC_COPY
// prototype (from hipcc diagnostic): (v4i addrspace(1)*, v4i addrspace(3)*, imm, imm)
typedef __attribute__((address_space(1))) v4i* g_v4i_p;
typedef __attribute__((address_space(3))) v4i* l_v4i_p;
__device__ __forceinline__ void copy16_g2l(const float* __restrict__ g, float* l) {
  // emits global_load_async_to_lds_b128 (tracked by ASYNCcnt)
  __builtin_amdgcn_global_load_async_to_lds_b128((g_v4i_p)g, (l_v4i_p)l, 0, 0);
}
__device__ __forceinline__ void wait_async_copies() {
  __builtin_amdgcn_s_wait_asynccnt(0);
}
#else
__device__ __forceinline__ void copy16_g2l(const float* __restrict__ g, float* l) {
  *(v4f*)l = *(const v4f*)g;   // synchronous b128 fallback
}
__device__ __forceinline__ void wait_async_copies() {}
#endif

// ---------------- WMMA helpers ----------------
__device__ __forceinline__ v8f wmma_bf16(v16bf a, v16bf b, v8f c) {
  // D = A(16x32 bf16) * B(32x16 bf16) + C(16x16 f32)
  return __builtin_amdgcn_wmma_f32_16x16x32_bf16(
      /*neg_a=*/false, a, /*neg_b=*/false, b,
      /*c_mod=*/(short)0, c, /*reuse_a=*/false, /*reuse_b=*/false);
}

// A-fragment (16x32, 16-bit): lane<16 -> row=lane, K = 0..15;
// lane>=16 -> row=lane-16, K = 16..31.  src is row-major [row][ldk].
__device__ __forceinline__ v16bf load_a_frag(const bf16* src, int row_base, int ldk) {
  int lane = threadIdx.x & 31;
  int row  = row_base + (lane & 15);
  int k0   = (lane < 16) ? 0 : 16;
  const bf16* p = src + row * ldk + k0;
  v16bf a;
#pragma unroll
  for (int i = 0; i < 8; ++i) a[i] = p[i];
#pragma unroll
  for (int i = 0; i < 8; ++i) a[8 + i] = p[8 + i];
  return a;
}

// B-fragment (32x16, 16-bit): lane<16 -> col=lane, K=0..15;
// lane>=16 -> col=lane-16, K=16..31.  src is col-major-of-B: [col][ldk].
__device__ __forceinline__ v16bf load_b_frag(const bf16* src, int col_base, int ldk) {
  int lane = threadIdx.x & 31;
  int col  = col_base + (lane & 15);
  int k0   = (lane < 16) ? 0 : 16;
  const bf16* p = src + col * ldk + k0;
  v16bf b;
#pragma unroll
  for (int i = 0; i < 16; ++i) b[i] = p[i];
  return b;
}

__device__ __forceinline__ float rowmax16(float v) {
#pragma unroll
  for (int o = 8; o >= 1; o >>= 1) v = fmaxf(v, __shfl_xor(v, o, 16));
  return v;
}
__device__ __forceinline__ float rowsum16(float v) {
#pragma unroll
  for (int o = 8; o >= 1; o >>= 1) v += __shfl_xor(v, o, 16);
  return v;
}

// ---------------- GEMM: C[M,N] (+= residual) = A[M,K] * B[K,N] ----------------
// fp32 in HBM, bf16 WMMA. Block: 256 threads (8 waves). Tile: BM=128, BN=64, BK=32.
// Double-buffered f32 staging via async global->LDS DMA, then LDS-local f32->bf16
// conversion pass (B transposed to [n][k]); wave w computes rows [16w,16w+16).
template <bool RESIDUAL>
__global__ __launch_bounds__(256)
void gemm_bf16_kernel(const float* __restrict__ A, const float* __restrict__ B,
                      float* __restrict__ C, int M, int N, int K) {
  __shared__ float Af[2][128 * 32];  // staged A tile (f32, [m][k])
  __shared__ float Bf[2][32 * 64];   // staged B tile (f32, [k][n])
  __shared__ bf16  As[128 * 32];     // [m][k] bf16
  __shared__ bf16  Bs[64 * 32];      // [n][k] bf16 (transposed)

  const int n0 = blockIdx.x * 64;
  const int m0 = blockIdx.y * 128;
  const int tid  = threadIdx.x;
  const int wave = tid >> 5;
  const int lane = tid & 31;

  auto stage = [&](int kt, int buf) {
    // A tile: 1024 v4f, 4 per thread
#pragma unroll
    for (int i = 0; i < 4; ++i) {
      int idx4 = tid + i * 256;
      int m = idx4 >> 3, kq = idx4 & 7;
      copy16_g2l(A + (size_t)(m0 + m) * K + kt + kq * 4, &Af[buf][idx4 * 4]);
    }
    // B tile: 512 v4f, 2 per thread
#pragma unroll
    for (int i = 0; i < 2; ++i) {
      int idx4 = tid + i * 256;
      int k = idx4 >> 4, nq = idx4 & 15;
      copy16_g2l(B + (size_t)(kt + k) * N + n0 + nq * 4, &Bf[buf][idx4 * 4]);
    }
  };

  v8f acc[4] = {v8f{}, v8f{}, v8f{}, v8f{}};

  stage(0, 0);
  int cur = 0;
  for (int kt = 0; kt < K; kt += 32) {
    wait_async_copies();
    __syncthreads();   // staging of buf[cur] visible; prev compute done with As/Bs

    // convert f32 staging -> bf16 fragments layout
#pragma unroll
    for (int i = 0; i < 16; ++i) {
      int idx = tid + i * 256;
      As[idx] = (bf16)Af[cur][idx];
    }
#pragma unroll
    for (int i = 0; i < 8; ++i) {
      int idx = tid + i * 256;
      int n = idx & 63, k = idx >> 6;
      Bs[n * 32 + k] = (bf16)Bf[cur][k * 64 + n];
    }
    __syncthreads();

    // overlap: DMA next chunk while this chunk's WMMAs run
    if (kt + 32 < K) stage(kt + 32, cur ^ 1);

    v16bf af = load_a_frag(As, wave * 16, 32);
#pragma unroll
    for (int nt = 0; nt < 4; ++nt) {
      v16bf bfr = load_b_frag(Bs, nt * 16, 32);
      acc[nt] = wmma_bf16(af, bfr, acc[nt]);
    }
    cur ^= 1;
  }

  // C fragment: VGPR r -> row = r + 8*(lane>=16); col = lane&15
  const int row_off = (lane >> 4) * 8;
  const int col_in  = lane & 15;
#pragma unroll
  for (int nt = 0; nt < 4; ++nt) {
#pragma unroll
    for (int r = 0; r < 8; ++r) {
      size_t m = (size_t)(m0 + wave * 16 + row_off + r);
      size_t n = (size_t)(n0 + nt * 16 + col_in);
      float v = acc[nt][r];
      if (RESIDUAL) v += C[m * N + n];
      C[m * N + n] = v;
    }
  }
}

// ---------------- flash attention ----------------
// q:   [S][H][192] f32 (rope applied), k: [S][H][192] f32, kvb: [S][H][256] (v = last 128)
// out: [S][H][128] f32.  grid: (S/64, H); block 128 threads = 4 waves, 16 q-rows/wave.
__global__ __launch_bounds__(128)
void flash_attn_kernel(const float* __restrict__ q, const float* __restrict__ k,
                       const float* __restrict__ kvb, float* __restrict__ out) {
  __shared__ bf16 Qs[64 * 192];    // [qrow][d]
  __shared__ bf16 Ks[32 * 192];    // [key][d]   (== B-fragment layout for Q*K^T)
  __shared__ bf16 Vs[128 * 32];    // [vdim][key] (B-fragment layout for P*V)
  __shared__ bf16 Ps[4 * 16 * 32]; // per-wave P bounce [16][32]

  const int q0   = blockIdx.x * 64;
  const int head = blockIdx.y;
  const int tid  = threadIdx.x;
  const int wave = tid >> 5;
  const int lane = tid & 31;

  // stage Q tile (3072 v4f / 128 thr = 24 each)
#pragma unroll
  for (int i = 0; i < 24; ++i) {
    int idx4 = tid + i * 128;
    int r = idx4 / 48, dq = idx4 % 48;
    v4f v = *(const v4f*)(q + ((size_t)(q0 + r) * NHEAD + head) * QKD + dq * 4);
#pragma unroll
    for (int j = 0; j < 4; ++j) Qs[r * 192 + dq * 4 + j] = (bf16)v[j];
  }
  __syncthreads();

  // preload Q fragments (6 chunks of K=32 over d=192)
  v16bf qf[6];
#pragma unroll
  for (int c = 0; c < 6; ++c) qf[c] = load_a_frag(Qs + c * 32, wave * 16, 192);

  const float NEG_INF = -__builtin_inff();
  float m_r[8], l_r[8];
  v8f o_acc[8];
#pragma unroll
  for (int r = 0; r < 8; ++r) { m_r[r] = NEG_INF; l_r[r] = 0.f; }
#pragma unroll
  for (int t = 0; t < 8; ++t) o_acc[t] = v8f{};

  const int row_off = (lane >> 4) * 8;
  const int col_in  = lane & 15;
  const int kend = q0 + 64;  // causal limit for this query tile

  for (int kt0 = 0; kt0 < kend; kt0 += 32) {
    __syncthreads();  // previous iter's Ks/Vs/Ps reads complete
    // stage K tile (1536 v4f / 128 = 12 each), row-major == B layout
#pragma unroll
    for (int i = 0; i < 12; ++i) {
      int idx4 = tid + i * 128;
      int kk = idx4 / 48, dq = idx4 % 48;
      v4f v = *(const v4f*)(k + ((size_t)(kt0 + kk) * NHEAD + head) * QKD + dq * 4);
#pragma unroll
      for (int j = 0; j < 4; ++j) Ks[kk * 192 + dq * 4 + j] = (bf16)v[j];
    }
    // stage V tile transposed [vdim][key] (1024 v4f / 128 = 8 each)
#pragma unroll
    for (int i = 0; i < 8; ++i) {
      int idx4 = tid + i * 128;
      int kk = idx4 >> 5, dq = idx4 & 31;
      v4f v = *(const v4f*)(kvb + ((size_t)(kt0 + kk) * NHEAD + head) * (DN + DV) + DN + dq * 4);
#pragma unroll
      for (int j = 0; j < 4; ++j) Vs[(dq * 4 + j) * 32 + kk] = (bf16)v[j];
    }
    __syncthreads();

    // scores S[16 x 32] for this wave: two 16x16 tiles, contraction d=192
    v8f s0 = v8f{}, s1 = v8f{};
#pragma unroll
    for (int c = 0; c < 6; ++c) {
      v16bf b0 = load_b_frag(Ks + c * 32, 0, 192);
      s0 = wmma_bf16(qf[c], b0, s0);
      v16bf b1 = load_b_frag(Ks + c * 32, 16, 192);
      s1 = wmma_bf16(qf[c], b1, s1);
    }

    // scale + causal mask
    float sc0[8], sc1[8];
#pragma unroll
    for (int r = 0; r < 8; ++r) {
      int qi = q0 + wave * 16 + row_off + r;
      sc0[r] = (kt0 + col_in      <= qi) ? s0[r] * ATTN_SCALE : NEG_INF;
      sc1[r] = (kt0 + 16 + col_in <= qi) ? s1[r] * ATTN_SCALE : NEG_INF;
    }

    // online softmax update
    float p0[8], p1[8];
#pragma unroll
    for (int r = 0; r < 8; ++r) {
      float mx = rowmax16(fmaxf(sc0[r], sc1[r]));
      float mn = fmaxf(m_r[r], mx);
      float alpha = __expf(m_r[r] - mn);   // first pass: exp(-inf) = 0
      p0[r] = __expf(sc0[r] - mn);
      p1[r] = __expf(sc1[r] - mn);
      float rs = rowsum16(p0[r] + p1[r]);
      l_r[r] = l_r[r] * alpha + rs;
      m_r[r] = mn;
#pragma unroll
      for (int t = 0; t < 8; ++t) o_acc[t][r] *= alpha;
    }

    // rearrange P from C-layout to A-layout through per-wave LDS bounce
    bf16* Pw = Ps + wave * 16 * 32;
#pragma unroll
    for (int r = 0; r < 8; ++r) {
      Pw[(row_off + r) * 32 + col_in]      = (bf16)p0[r];
      Pw[(row_off + r) * 32 + 16 + col_in] = (bf16)p1[r];
    }
    __syncthreads();

    v16bf pf = load_a_frag(Pw, 0, 32);
#pragma unroll
    for (int t = 0; t < 8; ++t) {
      v16bf vb = load_b_frag(Vs, t * 16, 32);
      o_acc[t] = wmma_bf16(pf, vb, o_acc[t]);
    }
  }

  // normalize and write out [S][H][128]
#pragma unroll
  for (int t = 0; t < 8; ++t) {
#pragma unroll
    for (int r = 0; r < 8; ++r) {
      size_t row = (size_t)(q0 + wave * 16 + row_off + r);
      out[(row * NHEAD + head) * DV + t * 16 + col_in] = o_acc[t][r] / l_r[r];
    }
  }
}

// ---------------- small kernels ----------------
__global__ void embed_kernel(const int* __restrict__ tokens, const float* __restrict__ emb,
                             float* __restrict__ h) {
  int s = blockIdx.x;
  int t = tokens[s];
  for (int i = threadIdx.x; i < DIM; i += blockDim.x)
    h[(size_t)s * DIM + i] = emb[(size_t)t * DIM + i];
}

__global__ void rmsnorm_kernel(const float* __restrict__ in, const float* __restrict__ w,
                               float* __restrict__ out, int width, int in_stride, int out_stride) {
  int row = blockIdx.x;
  const float* ip = in + (size_t)row * in_stride;
  float ss = 0.f;
  for (int i = threadIdx.x; i < width; i += blockDim.x) { float v = ip[i]; ss += v * v; }
#pragma unroll
  for (int o = 16; o >= 1; o >>= 1) ss += __shfl_xor(ss, o, 32);
  __shared__ float red[8];
  int wave = threadIdx.x >> 5, lane = threadIdx.x & 31;
  if (lane == 0) red[wave] = ss;
  __syncthreads();
  if (wave == 0) {
    float v = (lane < 8) ? red[lane] : 0.f;
#pragma unroll
    for (int o = 4; o >= 1; o >>= 1) v += __shfl_xor(v, o, 32);
    if (lane == 0) red[0] = v;
  }
  __syncthreads();
  float inv = rsqrtf(red[0] / (float)width + 1e-6f);
  float* op = out + (size_t)row * out_stride;
  for (int i = threadIdx.x; i < width; i += blockDim.x) op[i] = ip[i] * inv * w[i];
}

// rope on q's last 64 dims per head, interleaved pairs
__global__ void rope_q_kernel(float* __restrict__ q, const float* __restrict__ cb,
                              const float* __restrict__ sb) {
  int idx = blockIdx.x * blockDim.x + threadIdx.x;  // S*H*32
  if (idx >= S_LEN * NHEAD * (DR / 2)) return;
  int j = idx & 31;
  int sh = idx >> 5;
  int head = sh & (NHEAD - 1);
  int s = sh >> 4;
  float c = cb[s * 32 + j], sn = sb[s * 32 + j];
  float* p = q + ((size_t)s * NHEAD + head) * QKD + DN + 2 * j;
  float x0 = p[0], x1 = p[1];
  p[0] = x0 * c - x1 * sn;
  p[1] = x0 * sn + x1 * c;
}

// assemble k_all[s][h][0:128] = k_nope, [128:192] = rope(k_pe) broadcast over heads
__global__ void assemble_k_kernel(const float* __restrict__ kvb, const float* __restrict__ kv_all,
                                  const float* __restrict__ cb, const float* __restrict__ sb,
                                  float* __restrict__ k_all) {
  int s = blockIdx.x;
  for (int i = threadIdx.x; i < NHEAD * QKD; i += blockDim.x) {
    int head = i / QKD, d = i % QKD;
    float v;
    if (d < DN) {
      v = kvb[((size_t)s * NHEAD + head) * (DN + DV) + d];
    } else {
      int dd = d - DN;
      int j = dd >> 1;
      float x0 = kv_all[(size_t)s * (KVR + DR) + KVR + 2 * j];
      float x1 = kv_all[(size_t)s * (KVR + DR) + KVR + 2 * j + 1];
      float c = cb[s * 32 + j], sn = sb[s * 32 + j];
      v = (dd & 1) ? (x0 * sn + x1 * c) : (x0 * c - x1 * sn);
    }
    k_all[(size_t)s * NHEAD * QKD + i] = v;
  }
}

__global__ void silu_gate_kernel(float* __restrict__ a, const float* __restrict__ b, size_t n) {
  size_t i = (size_t)blockIdx.x * blockDim.x + threadIdx.x;
  if (i < n) {
    float x = a[i];
    a[i] = (x / (1.f + __expf(-x))) * b[i];
  }
}

// ---------------- host orchestration ----------------
static void launch_gemm(const float* A, const float* B, float* C, int M, int N, int K,
                        bool residual, hipStream_t stream) {
  dim3 grid(N / 64, M / 128), block(256);
  if (residual)
    gemm_bf16_kernel<true><<<grid, block, 0, stream>>>(A, B, C, M, N, K);
  else
    gemm_bf16_kernel<false><<<grid, block, 0, stream>>>(A, B, C, M, N, K);
}

extern "C" void kernel_launch(void* const* d_in, const int* in_sizes, int n_in,
                              void* d_out, int out_size, void* d_ws, size_t ws_size,
                              hipStream_t stream) {
  (void)in_sizes; (void)n_in; (void)out_size; (void)ws_size;

  const int*   tokens      = (const int*)d_in[0];
  const float* embedw      = (const float*)d_in[1];
  const float* attn_norm_w = (const float*)d_in[2];
  const float* wq          = (const float*)d_in[3];
  const float* wkv_a       = (const float*)d_in[4];
  const float* kv_norm_w   = (const float*)d_in[5];
  const float* wkv_b       = (const float*)d_in[6];
  const float* wo          = (const float*)d_in[7];
  const float* ffn_norm_w  = (const float*)d_in[8];
  const float* w1          = (const float*)d_in[9];
  const float* w2          = (const float*)d_in[10];
  const float* w3          = (const float*)d_in[11];
  const float* norm_w      = (const float*)d_in[12];
  const float* head_w      = (const float*)d_in[13];
  const float* fcos        = (const float*)d_in[14];
  const float* fsin        = (const float*)d_in[15];
  float* out = (float*)d_out;

  // workspace layout (floats)
  float* ws = (float*)d_ws;
  float* h        = ws;                               // S*DIM
  float* x        = h        + (size_t)S_LEN * DIM;   // S*DIM
  float* qbuf     = x        + (size_t)S_LEN * DIM;   // S*H*QKD
  float* kv_all   = qbuf     + (size_t)S_LEN * NHEAD * QKD;       // S*576
  float* kv_norm  = kv_all   + (size_t)S_LEN * (KVR + DR);        // S*512
  float* kvb      = kv_norm  + (size_t)S_LEN * KVR;               // S*H*256
  float* k_all    = kvb      + (size_t)S_LEN * NHEAD * (DN + DV); // S*H*192
  float* attn_out = k_all    + (size_t)S_LEN * NHEAD * QKD;       // S*H*128
  float* ffn1     = attn_out + (size_t)S_LEN * NHEAD * DV;        // S*INTER
  float* ffn3     = ffn1     + (size_t)S_LEN * INTER;             // S*INTER

  dim3 b256(256);

  embed_kernel<<<S_LEN, b256, 0, stream>>>(tokens, embedw, h);

  for (int l = 0; l < NLAYER; ++l) {
    const float* wq_l   = wq    + (size_t)l * DIM * NHEAD * QKD;
    const float* wkva_l = wkv_a + (size_t)l * DIM * (KVR + DR);
    const float* wkvb_l = wkv_b + (size_t)l * KVR * NHEAD * (DN + DV);
    const float* wo_l   = wo    + (size_t)l * NHEAD * DV * DIM;
    const float* w1_l   = w1    + (size_t)l * DIM * INTER;
    const float* w2_l   = w2    + (size_t)l * INTER * DIM;
    const float* w3_l   = w3    + (size_t)l * DIM * INTER;

    // attention
    rmsnorm_kernel<<<S_LEN, b256, 0, stream>>>(h, attn_norm_w + (size_t)l * DIM, x, DIM, DIM, DIM);
    launch_gemm(x, wq_l, qbuf, S_LEN, NHEAD * QKD, DIM, false, stream);
    rope_q_kernel<<<(S_LEN * NHEAD * (DR / 2)) / 256, b256, 0, stream>>>(qbuf, fcos, fsin);
    launch_gemm(x, wkva_l, kv_all, S_LEN, KVR + DR, DIM, false, stream);
    rmsnorm_kernel<<<S_LEN, b256, 0, stream>>>(kv_all, kv_norm_w + (size_t)l * KVR, kv_norm,
                                               KVR, KVR + DR, KVR);
    launch_gemm(kv_norm, wkvb_l, kvb, S_LEN, NHEAD * (DN + DV), KVR, false, stream);
    assemble_k_kernel<<<S_LEN, b256, 0, stream>>>(kvb, kv_all, fcos, fsin, k_all);
    {
      dim3 grid(S_LEN / 64, NHEAD), block(128);
      flash_attn_kernel<<<grid, block, 0, stream>>>(qbuf, k_all, kvb, attn_out);
    }
    launch_gemm(attn_out, wo_l, h, S_LEN, DIM, NHEAD * DV, true, stream);  // h += attn @ wo

    // FFN
    rmsnorm_kernel<<<S_LEN, b256, 0, stream>>>(h, ffn_norm_w + (size_t)l * DIM, x, DIM, DIM, DIM);
    launch_gemm(x, w1_l, ffn1, S_LEN, INTER, DIM, false, stream);
    launch_gemm(x, w3_l, ffn3, S_LEN, INTER, DIM, false, stream);
    {
      size_t n = (size_t)S_LEN * INTER;
      silu_gate_kernel<<<(unsigned)((n + 255) / 256), b256, 0, stream>>>(ffn1, ffn3, n);
    }
    launch_gemm(ffn1, w2_l, h, S_LEN, DIM, INTER, true, stream);  // h += gate @ w2
  }

  // final norm + vocab head
  rmsnorm_kernel<<<S_LEN, b256, 0, stream>>>(h, norm_w, x, DIM, DIM, DIM);
  launch_gemm(x, head_w, out, S_LEN, VOCAB, DIM, false, stream);
}
